// Akscore2_DockS_33964601377159
// MI455X (gfx1250) — matrix operand order repaired and checked by reading, not compiled, not run on hardware
//
#include <hip/hip_runtime.h>
#include <stdint.h>

#define NN 50000          // nodes
#define NE 1600000        // real edges
#define E2T (NE + NN)     // edges + self loops
#define HID 128
#define EDIM 24
#define NL 5
#define NG 128
#define NDIM 73

typedef __attribute__((ext_vector_type(16))) __bf16 bf16x16;
typedef __attribute__((ext_vector_type(2)))  __bf16 bf16x2;
typedef __attribute__((ext_vector_type(8)))  float  f32x8;
typedef int v4i __attribute__((vector_size(16)));

union BFrag { uint32_t u[8]; bf16x16 v; };

#if defined(__has_builtin)
# if __has_builtin(__builtin_amdgcn_global_load_async_to_lds_b128)
#  define HAS_ASYNC_LDS 1
# endif
# if __has_builtin(__builtin_amdgcn_s_wait_asynccnt)
#  define HAS_WAIT_ASYNC 1
# endif
# if __has_builtin(__builtin_amdgcn_cvt_pk_bf16_f32)
#  define HAS_CVT_PK_BF16 1
# endif
#endif

__device__ __forceinline__ uint32_t f2bf(float f) {
  uint32_t u = __float_as_uint(f);
  return (u + 0x7FFFu + ((u >> 16) & 1u)) >> 16;   // RNE truncate to bf16
}
__device__ __forceinline__ uint32_t pack_bf16(float a, float b) {
#if defined(HAS_CVT_PK_BF16)
  union { bf16x2 v; uint32_t u; } r;
  r.v = __builtin_amdgcn_cvt_pk_bf16_f32(a, b);
  return r.u;
#else
  return f2bf(a) | (f2bf(b) << 16);
#endif
}
__device__ __forceinline__ void atomicMaxF(float* addr, float v) {
  if (v >= 0.f) atomicMax((int*)addr, __float_as_int(v));
  else          atomicMin((unsigned int*)addr, (unsigned int)__float_as_int(v));
}
__device__ __forceinline__ void atomAddF(float* p, float v) {
  unsafeAtomicAdd(p, v);   // -> global_atomic_add_f32
}

// ---------------- generic zero ----------------
__global__ void k_zero(float* p, int n) {
  int i = blockIdx.x * blockDim.x + threadIdx.x;
  if (i < n) p[i] = 0.f;
}

// ---------------- degree + edge-attr segment sum over dst ----------------
__global__ void k_deg_sumea(const int* __restrict__ ei, const float* __restrict__ ea,
                            float* __restrict__ deg, float* __restrict__ sum_ea) {
  int e = blockIdx.x * blockDim.x + threadIdx.x;
  if (e >= NE) return;
  int d = ei[NE + e];
  atomAddF(&deg[d], 1.0f);
  const float* ep = ea + (long long)e * EDIM;
  float* sp = sum_ea + (long long)d * EDIM;
  #pragma unroll
  for (int k = 0; k < EDIM; ++k) atomAddF(&sp[k], ep[k]);
}

// loop_attr = sum_ea / max(deg,1)   (in place)
__global__ void k_loop_attr(float* sum_ea, const float* deg) {
  int i = blockIdx.x * blockDim.x + threadIdx.x;
  if (i >= NN * EDIM) return;
  sum_ea[i] = sum_ea[i] / fmaxf(deg[i / EDIM], 1.0f);
}

// ---------------- input projection h = x @ W_in + b_in ----------------
__global__ void k_input_proj(const float* __restrict__ x, const float* __restrict__ W,
                             const float* __restrict__ b, float* __restrict__ h) {
  __shared__ float xs[NDIM];
  int n = blockIdx.x, j = threadIdx.x;
  if (j < NDIM) xs[j] = x[(long long)n * NDIM + j];
  __syncthreads();
  float acc = b[j];
  #pragma unroll 1
  for (int k = 0; k < NDIM; ++k) acc += xs[k] * W[k * HID + j];
  h[(long long)n * HID + j] = acc;
}

// ---------------- h (f32) -> packed bf16 rows (64 dwords per row) ----------------
__global__ void k_h2bf(const float* __restrict__ h, uint32_t* __restrict__ hbf) {
  int i = blockIdx.x * blockDim.x + threadIdx.x;
  if (i >= NN * (HID / 2)) return;
  hbf[i] = pack_bf16(h[2 * i], h[2 * i + 1]);
}

// ---------------- pack Wl/Wr into WMMA B-fragment layout (bf16) ----------------
// packed[which][kt(4)][nt(8)][lane(32)][8 dwords]
__global__ void k_pack_weights(const float* __restrict__ Wl, const float* __restrict__ Wr,
                               uint32_t* __restrict__ packed) {
  int lane = threadIdx.x;
  int kt = blockIdx.x >> 3;
  int nt = blockIdx.x & 7;
  int which = blockIdx.y;
  const float* W = which ? Wr : Wl;
  int n  = nt * 16 + (lane & 15);
  int k0 = kt * 32 + ((lane >> 4) ? 8 : 0);
  uint32_t* dst = packed + (size_t)which * 8192 + (((kt * 8 + nt) * 32 + lane) * 8);
  #pragma unroll
  for (int d = 0; d < 8; ++d)
    dst[d] = pack_bf16(W[(k0 + 2 * d + (d >= 4 ? 8 : 0)) * HID + n], W[(k0 + 2 * d + (d >= 4 ? 8 : 0) + 1) * HID + n]);
}

// ---------------- xl = h@Wl+bl and xr = h@Wr+br, fused, bf16 WMMA ----------------
// block = 256 (8 waves). LDS holds both packed weight matrices (64 KB), staged
// with GLOBAL_LOAD_ASYNC_TO_LDS_B128 (ASYNCcnt) when available.
__global__ void __launch_bounds__(256) k_gemm_nodes(
    const uint32_t* __restrict__ hbf, const uint32_t* __restrict__ packed,
    const float* __restrict__ bl, const float* __restrict__ br,
    float* __restrict__ xl, float* __restrict__ xr) {
  __shared__ uint32_t Bs[16384];   // [which][kt][nt][lane][8]
  int tid = threadIdx.x;
  int mt  = blockIdx.x;

#if defined(HAS_ASYNC_LDS)
  #pragma unroll
  for (int i = 0; i < 16; ++i) {
    int idx = (i * 256 + tid) * 4;
    __builtin_amdgcn_global_load_async_to_lds_b128((v4i*)(void*)(packed + idx),
                                                   (v4i*)(void*)(Bs + idx), 0, 0);
  }
# if defined(HAS_WAIT_ASYNC)
  __builtin_amdgcn_s_wait_asynccnt(0);
# else
  asm volatile("s_wait_asynccnt 0x0" ::: "memory");
# endif
#else
  for (int i = tid; i < 16384; i += 256) Bs[i] = packed[i];
#endif
  __syncthreads();

  int wave = tid >> 5, lane = tid & 31;
  int nt = wave;                       // 8 waves -> 8 col tiles
  int half = lane >> 4, l15 = lane & 15;
  f32x8 cl = {0.f, 0.f, 0.f, 0.f, 0.f, 0.f, 0.f, 0.f};
  f32x8 cr = {0.f, 0.f, 0.f, 0.f, 0.f, 0.f, 0.f, 0.f};
  const uint32_t* arow = hbf + (size_t)(mt * 16 + l15) * (HID / 2) + half * 4;

  #pragma unroll
  for (int kt = 0; kt < 4; ++kt) {
    BFrag a, b0, b1;
    const uint32_t* ap = arow + kt * 16;
    #pragma unroll
    for (int d = 0; d < 4; ++d) { a.u[d] = ap[d]; a.u[4 + d] = ap[8 + d]; }
    const uint32_t* bp = Bs + (((kt * 8 + nt) * 32 + lane) * 8);
    #pragma unroll
    for (int d = 0; d < 8; ++d) { b0.u[d] = bp[d]; b1.u[d] = bp[8192 + d]; }
    cl = __builtin_amdgcn_wmma_f32_16x16x32_bf16(false, a.v, false, b0.v, (short)0, cl,
                                                 false, false);
    cr = __builtin_amdgcn_wmma_f32_16x16x32_bf16(false, a.v, false, b1.v, (short)0, cr,
                                                 false, false);
  }
  int n = nt * 16 + l15;
  int mbase = mt * 16 + half * 8;
  float bvl = bl[n], bvr = br[n];
  #pragma unroll
  for (int v = 0; v < 8; ++v) {
    xl[(size_t)(mbase + v) * HID + n] = cl[v] + bvl;
    xr[(size_t)(mbase + v) * HID + n] = cr[v] + bvr;
  }
}

// ---------------- per-layer init: h = bg ; lmax=-inf ; denom=0 ----------------
__global__ void k_init_layer(float* __restrict__ h, const float* __restrict__ bg,
                             float* __restrict__ lmax, float* __restrict__ denom) {
  int i = blockIdx.x * blockDim.x + threadIdx.x;
  if (i >= NN * HID) return;
  h[i] = bg[i & (HID - 1)];
  if (i < NN) { lmax[i] = __int_as_float(0xFF800000); denom[i] = 0.f; }
}

// ---------------- edge logits + atomic segment max ----------------
#define EPW 8   // edges per wave
__global__ void k_edge_logit(const int* __restrict__ ei, const float* __restrict__ ea,
                             const float* __restrict__ loop_attr,
                             const float* __restrict__ xl, const float* __restrict__ xr,
                             const float* __restrict__ We, const float* __restrict__ att,
                             float* __restrict__ logit, float* __restrict__ lmax) {
  __shared__ float Ws[EDIM * HID];
  for (int t = threadIdx.x; t < EDIM * HID; t += blockDim.x) Ws[t] = We[t];
  __syncthreads();
  int lane = threadIdx.x & 31;
  int wave = threadIdx.x >> 5;
  float a0 = att[lane], a1 = att[lane + 32], a2 = att[lane + 64], a3 = att[lane + 96];
  int base = (blockIdx.x * (blockDim.x >> 5) + wave) * EPW;
  for (int it = 0; it < EPW; ++it) {
    int e2 = base + it;
    if (e2 >= E2T) return;
    int s, d; const float* ep;
    if (e2 < NE) { s = ei[e2]; d = ei[NE + e2]; ep = ea + (long long)e2 * EDIM; }
    else         { s = d = e2 - NE;             ep = loop_attr + (long long)s * EDIM; }
    const float* xls = xl + (long long)s * HID;
    const float* xrd = xr + (long long)d * HID;
    float m0 = xls[lane]      + xrd[lane];
    float m1 = xls[lane + 32] + xrd[lane + 32];
    float m2 = xls[lane + 64] + xrd[lane + 64];
    float m3 = xls[lane + 96] + xrd[lane + 96];
    #pragma unroll 1
    for (int k = 0; k < EDIM; ++k) {
      float ek = ep[k];
      const float* wr = Ws + k * HID;
      m0 += ek * wr[lane];      m1 += ek * wr[lane + 32];
      m2 += ek * wr[lane + 64]; m3 += ek * wr[lane + 96];
    }
    m0 = m0 > 0.f ? m0 : 0.2f * m0;  m1 = m1 > 0.f ? m1 : 0.2f * m1;
    m2 = m2 > 0.f ? m2 : 0.2f * m2;  m3 = m3 > 0.f ? m3 : 0.2f * m3;
    float sum = m0 * a0 + m1 * a1 + m2 * a2 + m3 * a3;
    #pragma unroll
    for (int off = 16; off > 0; off >>= 1) sum += __shfl_xor(sum, off, 32);
    if (lane == 0) { logit[e2] = sum; atomicMaxF(&lmax[d], sum); }
  }
}

// ---------------- ex = exp(logit - lmax[dst]); denom += ex (logit reused as ex) ----------------
__global__ void k_edge_exp(const int* __restrict__ ei, float* __restrict__ logit,
                           const float* __restrict__ lmax, float* __restrict__ denom) {
  int e2 = blockIdx.x * blockDim.x + threadIdx.x;
  if (e2 >= E2T) return;
  int d = (e2 < NE) ? ei[NE + e2] : (e2 - NE);
  float ex = __expf(logit[e2] - lmax[d]);
  logit[e2] = ex;
  atomAddF(&denom[d], ex);
}

// ---------------- h[dst] += alpha * xl[src] (wave per edge) ----------------
__global__ void k_edge_aggr(const int* __restrict__ ei, const float* __restrict__ exb,
                            const float* __restrict__ denom, const float* __restrict__ xl,
                            float* __restrict__ h) {
  int gid = blockIdx.x * blockDim.x + threadIdx.x;
  int e2 = gid >> 5, lane = gid & 31;
  if (e2 >= E2T) return;
  int s, d;
  if (e2 < NE) { s = ei[e2]; d = ei[NE + e2]; } else { s = d = e2 - NE; }
  float alpha = exb[e2] / denom[d];
  const float* xls = xl + (long long)s * HID;
  float* hd = h + (long long)d * HID;
  #pragma unroll
  for (int t = 0; t < 4; ++t)
    atomAddF(&hd[lane + 32 * t], alpha * xls[lane + 32 * t]);
}

__global__ void k_relu(float* h, int n) {
  int i = blockIdx.x * blockDim.x + threadIdx.x;
  if (i < n) h[i] = fmaxf(h[i], 0.f);
}

// ---------------- mean pool over graphs ----------------
__global__ void k_pool_accum(const int* __restrict__ batch, const float* __restrict__ h,
                             float* __restrict__ pooled, float* __restrict__ cnt) {
  int gid = blockIdx.x * blockDim.x + threadIdx.x;
  int n = gid >> 5, lane = gid & 31;
  if (n >= NN) return;
  int b = batch[n];
  const float* hn = h + (long long)n * HID;
  float* pg = pooled + (long long)b * HID;
  #pragma unroll
  for (int t = 0; t < 4; ++t) atomAddF(&pg[lane + 32 * t], hn[lane + 32 * t]);
  if (lane == 0) atomAddF(&cnt[b], 1.0f);
}
__global__ void k_pool_div(float* pooled, const float* cnt) {
  int i = blockIdx.x * blockDim.x + threadIdx.x;
  if (i >= NG * HID) return;
  pooled[i] /= fmaxf(cnt[i >> 7], 1.0f);
}

// ---------------- small FC: out = [relu](in @ W + b) ----------------
__global__ void k_fc(const float* __restrict__ in, const float* __restrict__ W,
                     const float* __restrict__ b, float* __restrict__ out,
                     int G, int Kin, int Kout, int relu) {
  int i = blockIdx.x * blockDim.x + threadIdx.x;
  if (i >= G * Kout) return;
  int g = i / Kout, o = i % Kout;
  float acc = b[o];
  for (int k = 0; k < Kin; ++k) acc += in[g * Kin + k] * W[k * Kout + o];
  out[i] = relu ? fmaxf(acc, 0.f) : acc;
}

static inline char* bump(char*& p, size_t bytes) {
  char* r = p;
  p += (bytes + 255) & ~(size_t)255;
  return r;
}

extern "C" void kernel_launch(void* const* d_in, const int* in_sizes, int n_in,
                              void* d_out, int out_size, void* d_ws, size_t ws_size,
                              hipStream_t stream) {
  const float* x     = (const float*)d_in[0];
  const int*   ei    = (const int*)d_in[1];
  const float* ea    = (const float*)d_in[2];
  const int*   batch = (const int*)d_in[3];
  const float* W_in  = (const float*)d_in[4];
  const float* b_in  = (const float*)d_in[5];
  const float* Wl    = (const float*)d_in[6];
  const float* bl    = (const float*)d_in[7];
  const float* Wr    = (const float*)d_in[8];
  const float* br    = (const float*)d_in[9];
  const float* We    = (const float*)d_in[10];
  const float* att   = (const float*)d_in[11];
  const float* bg    = (const float*)d_in[12];
  const float* W1b = (const float*)d_in[13]; const float* b1b = (const float*)d_in[14];
  const float* W2b = (const float*)d_in[15]; const float* b2b = (const float*)d_in[16];
  const float* W3b = (const float*)d_in[17]; const float* b3b = (const float*)d_in[18];
  const float* W1r = (const float*)d_in[19]; const float* b1r = (const float*)d_in[20];
  const float* W2r = (const float*)d_in[21]; const float* b2r = (const float*)d_in[22];
  const float* W3r = (const float*)d_in[23]; const float* b3r = (const float*)d_in[24];
  float* out = (float*)d_out;

  char* p = (char*)d_ws;
  float*    h      = (float*)   bump(p, (size_t)NN * HID * 4);
  float*    xl     = (float*)   bump(p, (size_t)NN * HID * 4);
  float*    xr     = (float*)   bump(p, (size_t)NN * HID * 4);
  uint32_t* hbf    = (uint32_t*)bump(p, (size_t)NN * (HID / 2) * 4);
  float*    logit  = (float*)   bump(p, (size_t)E2T * 4);
  float*    lmax   = (float*)   bump(p, (size_t)NN * 4);
  float*    denom  = (float*)   bump(p, (size_t)NN * 4);
  float*    deg    = (float*)   bump(p, (size_t)NN * 4);
  float*    sum_ea = (float*)   bump(p, (size_t)NN * EDIM * 4);   // becomes loop_attr
  uint32_t* packed = (uint32_t*)bump(p, (size_t)2 * 8192 * 4);
  float*    pooled = (float*)   bump(p, (size_t)NG * HID * 4);
  float*    cnt    = (float*)   bump(p, (size_t)NG * 4);
  float*    t1     = (float*)   bump(p, (size_t)NG * HID * 4);
  float*    t2     = (float*)   bump(p, (size_t)NG * 64 * 4);

  // ---- precompute: degree, edge-attr mean per node (self-loop attrs) ----
  k_zero<<<(NN + 255) / 256, 256, 0, stream>>>(deg, NN);
  k_zero<<<(NN * EDIM + 255) / 256, 256, 0, stream>>>(sum_ea, NN * EDIM);
  k_deg_sumea<<<(NE + 255) / 256, 256, 0, stream>>>(ei, ea, deg, sum_ea);
  k_loop_attr<<<(NN * EDIM + 255) / 256, 256, 0, stream>>>(sum_ea, deg);

  // ---- input projection ----
  k_input_proj<<<NN, HID, 0, stream>>>(x, W_in, b_in, h);

  // ---- 5 GATv2 layers ----
  for (int L = 0; L < NL; ++L) {
    k_pack_weights<<<dim3(32, 2), 32, 0, stream>>>(Wl + (size_t)L * HID * HID,
                                                   Wr + (size_t)L * HID * HID, packed);
    k_h2bf<<<(NN * (HID / 2) + 255) / 256, 256, 0, stream>>>(h, hbf);
    k_gemm_nodes<<<NN / 16, 256, 0, stream>>>(hbf, packed,
                                              bl + L * HID, br + L * HID, xl, xr);
    k_init_layer<<<(NN * HID + 255) / 256, 256, 0, stream>>>(h, bg + L * HID, lmax, denom);
    k_edge_logit<<<(E2T + 8 * EPW - 1) / (8 * EPW), 256, 0, stream>>>(
        ei, ea, sum_ea, xl, xr, We + (size_t)L * EDIM * HID, att + L * HID, logit, lmax);
    k_edge_exp<<<(E2T + 255) / 256, 256, 0, stream>>>(ei, logit, lmax, denom);
    k_edge_aggr<<<((size_t)E2T * 32 + 255) / 256, 256, 0, stream>>>(ei, logit, denom, xl, h);
    k_relu<<<(NN * HID + 255) / 256, 256, 0, stream>>>(h, NN * HID);
  }

  // ---- mean pool ----
  k_zero<<<(NG * HID + 255) / 256, 256, 0, stream>>>(pooled, NG * HID);
  k_zero<<<1, 256, 0, stream>>>(cnt, NG);
  k_pool_accum<<<((size_t)NN * 32 + 255) / 256, 256, 0, stream>>>(batch, h, pooled, cnt);
  k_pool_div<<<(NG * HID + 255) / 256, 256, 0, stream>>>(pooled, cnt);

  // ---- MLP heads (bind -> out[0:128], rmsd -> out[128:256]) ----
  k_fc<<<(NG * HID + 255) / 256, 256, 0, stream>>>(pooled, W1b, b1b, t1, NG, HID, HID, 1);
  k_fc<<<(NG * 64 + 255) / 256, 256, 0, stream>>>(t1, W2b, b2b, t2, NG, HID, 64, 1);
  k_fc<<<1, 128, 0, stream>>>(t2, W3b, b3b, out, NG, 64, 1, 0);
  k_fc<<<(NG * HID + 255) / 256, 256, 0, stream>>>(pooled, W1r, b1r, t1, NG, HID, HID, 1);
  k_fc<<<(NG * 64 + 255) / 256, 256, 0, stream>>>(t1, W2r, b2r, t2, NG, HID, 64, 1);
  k_fc<<<1, 128, 0, stream>>>(t2, W3r, b3r, out + NG, NG, 64, 1, 0);
}
